// Attention_44074954391876
// MI455X (gfx1250) — compile-verified, matrix operand order checked by reading, not run on hardware
//
#include <hip/hip_runtime.h>
#include <hip/hip_bf16.h>

typedef __attribute__((ext_vector_type(16))) _Float16 v16h;
typedef __attribute__((ext_vector_type(8)))  _Float16 v8h;
typedef __attribute__((ext_vector_type(8)))  float    v8f;
typedef __attribute__((ext_vector_type(4)))  unsigned int u32x4;
typedef __attribute__((ext_vector_type(8)))  unsigned int u32x8;

#define DIMC    1024
#define HEADS   16
#define NSEQ    257
#define BATCH   64
#define M_TOTAL (BATCH * NSEQ)   // 16448
#define HD      64
#define NPAD    288              // 18 tiles of 16, 9 chunks of 32

__device__ __forceinline__ v8f zero8f() {
    v8f z;
#pragma unroll
    for (int i = 0; i < 8; ++i) z[i] = 0.f;
    return z;
}

__device__ __forceinline__ v8f wmma16(v16h a, v16h b, v8f c) {
    // D = A(16x32 f16) * B(32x16 f16) + C(16x16 f32)
    return __builtin_amdgcn_wmma_f32_16x16x32_f16(false, a, false, b, (short)0, c,
                                                  false, false);
}

// Build a 16x32 f16 A- or B-fragment per ISA layout:
// lane = hf*16 + r ; element i in [0,8): K = k0 + hf*8 + i
//                    element i in [8,16): K = k0 + 16 + hf*8 + (i-8)
// -> two contiguous 16B runs => 2x (ds|global)_load_b128 per fragment.
__device__ __forceinline__ v16h load_frag_h(const _Float16* base, int stride,
                                            int row, int k0, int hf) {
    const _Float16* p = base + (long)row * stride + k0 + hf * 8;
    v8h lo = *(const v8h*)(p);
    v8h hi = *(const v8h*)(p + 16);
    return __builtin_shufflevector(lo, hi, 0, 1, 2, 3, 4, 5, 6, 7,
                                   8, 9, 10, 11, 12, 13, 14, 15);
}

// ---------------------------------------------------------------------------
// TDM: DMA the 257x64 f16 K slab into LDS with hardware row padding
// (pad_interval = 32 DWORDs = one 128B row, pad_amount = 4 DWORDs = 8 halves
//  -> LDS row stride 72 halves). Descriptor per CDNA5 ISA D# groups 0/1.
// ---------------------------------------------------------------------------
__device__ __forceinline__ void tdm_load_K(const _Float16* kg, _Float16* Ksm) {
    const unsigned long long ga = (unsigned long long)(uintptr_t)kg;
    const unsigned int lds = (unsigned int)(uintptr_t)Ksm;  // addr[31:0] = LDS offset
    u32x4 g0;
    g0[0] = 1u;                                       // count=1, user mode
    g0[1] = lds;                                      // lds_addr
    g0[2] = (unsigned int)(ga & 0xFFFFFFFFu);         // global_addr[31:0]
    g0[3] = (unsigned int)((ga >> 32) & 0x01FFFFFFu)  // global_addr[56:32]
          | (2u << 30);                               // type = 2 ("image")
    u32x8 g1;
    g1[0] = (1u << 16)     // data_size = 2 bytes
          | (1u << 20)     // pad_enable
          | (4u << 22)     // pad_interval: 32 DWORDs
          | (3u << 25);    // pad_amount:   4 DWORDs
    g1[1] = (64u << 16);   // tensor_dim0 = 64 (bits 79:48)
    g1[2] = (257u << 16);  // tensor_dim1 = 257 (bits 111:80)
    g1[3] = (64u << 16);   // tile_dim0 = 64 (bits 127:112)
    g1[4] = 257u;          // tile_dim1 = 257 (bits 143:128), tile_dim2 = 0
    g1[5] = 64u;           // tensor_dim0_stride = 64 (bits 207:160)
    g1[6] = 0u;
    g1[7] = 0u;
    asm volatile("tensor_load_to_lds %0, %1" :: "s"(g0), "s"(g1) : "memory");
}

// ---------------------------------------------------------------------------
// Kernel 0: expand relative position bias table -> [H, N, N] f32
// ---------------------------------------------------------------------------
__global__ __launch_bounds__(256) void bias_expand_kernel(
    const float* __restrict__ table, const int* __restrict__ idx,
    float* __restrict__ biasTab) {
    const int i = blockIdx.x * 256 + threadIdx.x;
    if (i < NSEQ * NSEQ) {
        const int id = idx[i];
#pragma unroll
        for (int h = 0; h < HEADS; ++h)
            biasTab[(long)h * (NSEQ * NSEQ) + i] = table[id * HEADS + h];
    }
}

// ---------------------------------------------------------------------------
// Kernel 1: qkv = x @ qkv_w^T + bias, split/scale into q,k,v f16 slabs [B,H,N,d]
// block tile 128(M) x 64(J) x 32(K); 8 waves, each wave 32x32 via 2x2 WMMA
// (f32 inputs must be converted, so staging stays on the VGPR path here)
// ---------------------------------------------------------------------------
__global__ __launch_bounds__(256) void qkv_gemm_kernel(
    const float* __restrict__ x, const float* __restrict__ w,
    const float* __restrict__ qb, const float* __restrict__ vb,
    _Float16* __restrict__ qs, _Float16* __restrict__ ks,
    _Float16* __restrict__ vs) {
    __shared__ _Float16 As[128 * 40];  // 32 cols + 8 pad (keeps 16B alignment)
    __shared__ _Float16 Bs[64 * 40];

    const int tid = threadIdx.x;
    const int lane = tid & 31, wv = tid >> 5;
    const int wm = wv & 3, wn = wv >> 2;
    const int r = lane & 15, hf = lane >> 4;
    const int j0 = blockIdx.x * 64;
    const int m0 = blockIdx.y * 128;

    v8f acc00 = zero8f(), acc01 = zero8f(), acc10 = zero8f(), acc11 = zero8f();

    const int xr = tid >> 1, xc = (tid & 1) * 16;  // 128 rows x 32 cols of x
    const int wr = tid >> 2, wc = (tid & 3) * 8;   // 64 rows x 32 cols of w

    for (int kk0 = 0; kk0 < DIMC; kk0 += 32) {
        {
            const int m = m0 + xr;
            v8h h0, h1;
#pragma unroll
            for (int i = 0; i < 8; ++i) { h0[i] = (_Float16)0.f; h1[i] = (_Float16)0.f; }
            if (m < M_TOTAL) {
                const float* p = x + (long)m * DIMC + kk0 + xc;
#pragma unroll
                for (int i = 0; i < 8; ++i) {
                    h0[i] = (_Float16)p[i];
                    h1[i] = (_Float16)p[i + 8];
                }
            }
            *(v8h*)&As[xr * 40 + xc] = h0;
            *(v8h*)&As[xr * 40 + xc + 8] = h1;
        }
        {
            const float* p = w + (long)(j0 + wr) * DIMC + kk0 + wc;
            v8h hw;
#pragma unroll
            for (int i = 0; i < 8; ++i) hw[i] = (_Float16)p[i];
            *(v8h*)&Bs[wr * 40 + wc] = hw;
        }
        __syncthreads();

        v16h a0 = load_frag_h(As, 40, wm * 32 + r, 0, hf);
        v16h a1 = load_frag_h(As, 40, wm * 32 + 16 + r, 0, hf);
        v16h b0 = load_frag_h(Bs, 40, wn * 32 + r, 0, hf);
        v16h b1 = load_frag_h(Bs, 40, wn * 32 + 16 + r, 0, hf);
        acc00 = wmma16(a0, b0, acc00);
        acc01 = wmma16(a0, b1, acc01);
        acc10 = wmma16(a1, b0, acc10);
        acc11 = wmma16(a1, b1, acc11);
        __syncthreads();
    }

#pragma unroll
    for (int mi = 0; mi < 2; ++mi) {
#pragma unroll
        for (int ni = 0; ni < 2; ++ni) {
            v8f av = mi == 0 ? (ni == 0 ? acc00 : acc01)
                             : (ni == 0 ? acc10 : acc11);
            const int jj = j0 + wn * 32 + ni * 16 + r;  // C/D: N = lane%16
            const int cls = jj >> 10, c = jj & 1023;
            const int h = c >> 6, dd = c & 63;
            const float bias = (cls == 0) ? qb[c] : (cls == 2 ? vb[c] : 0.f);
            _Float16* slab = cls == 0 ? qs : (cls == 1 ? ks : vs);
#pragma unroll
            for (int e = 0; e < 8; ++e) {
                const int m = m0 + wm * 32 + mi * 16 + e + 8 * hf;  // M = e + 8*(lane/16)
                if (m < M_TOTAL) {
                    float v = av[e] + bias;
                    if (cls == 0) v *= 0.125f;  // scale = 64^-0.5 folded into q
                    const int b = m / NSEQ, n = m % NSEQ;
                    slab[(((long)b * HEADS + h) * NSEQ + n) * HD + dd] = (_Float16)v;
                }
            }
        }
    }
}

// ---------------------------------------------------------------------------
// Kernel 2: fused attention per (b,h). K via TDM DMA (hardware-padded rows),
// V transposed in LDS; per-wave 16-query tiles: QK^T WMMA -> bias/mask ->
// softmax -> PV WMMA.
// ---------------------------------------------------------------------------
__global__ __launch_bounds__(256) void attn_kernel(
    const _Float16* __restrict__ qs, const _Float16* __restrict__ ks,
    const _Float16* __restrict__ vs, const float* __restrict__ biasTab,
    const unsigned char* __restrict__ mask, _Float16* __restrict__ ao) {
    extern __shared__ char smem[];
    _Float16* Ksm = (_Float16*)smem;                 // [288][72] halves, 41472 B
    _Float16* Vtm = (_Float16*)(smem + 41472);       // [64][296] halves, 37888 B
    float* SC = (float*)(smem + 41472 + 37888);      // [8 waves][16][288] f32

    const int tid = threadIdx.x;
    const int lane = tid & 31, wv = tid >> 5;
    const int r = lane & 15, hf = lane >> 4;
    const int bh = blockIdx.x;
    const int b = bh >> 4, h = bh & 15;

    const _Float16* qg = qs + (long)bh * NSEQ * HD;
    const _Float16* kg = ks + (long)bh * NSEQ * HD;
    const _Float16* vg = vs + (long)bh * NSEQ * HD;

    // Stage K: one TDM tensor DMA (wave 0 only; TDM is a whole-wave op)
    if (wv == 0) {
        tdm_load_K(kg, Ksm);
    }
    // Zero-pad K rows 257..287 (31 rows x 64 halves) on the ds path
    for (int idx = tid; idx < (NPAD - NSEQ) * 8; idx += 256) {
        const int row = NSEQ + (idx >> 3), ch = (idx & 7) * 8;
        v8h z;
#pragma unroll
        for (int i = 0; i < 8; ++i) z[i] = (_Float16)0.f;
        *(v8h*)&Ksm[row * 72 + ch] = z;
    }
    // Stage V transposed: Vt[d][key] so PV B-fragments are contiguous reads
    for (int idx = tid; idx < NSEQ * HD; idx += 256) {
        const int key = idx >> 6, dd = idx & 63;
        Vtm[dd * 296 + key] = vg[key * HD + dd];
    }
    for (int idx = tid; idx < HD * (NPAD - NSEQ); idx += 256) {  // zero pad keys
        const int dd = idx / (NPAD - NSEQ), key = NSEQ + idx % (NPAD - NSEQ);
        Vtm[dd * 296 + key] = (_Float16)0.f;
    }
    if (wv == 0) {
        __builtin_amdgcn_s_wait_tensorcnt(0);  // K DMA complete
    }
    __syncthreads();

    const float* biasH = biasTab + (long)h * (NSEQ * NSEQ);
    const unsigned char* mb = mask + b * NSEQ;
    float* sc = SC + wv * (16 * NPAD);

    for (int qt = wv; qt < 17; qt += 8) {  // 17 query tiles over 8 waves
        const int qrow = qt * 16 + r;
        const int qr = qrow < NSEQ ? qrow : NSEQ - 1;  // clamp (stores guarded)
        v16h qa0 = load_frag_h(qg, HD, qr, 0, hf);
        v16h qa1 = load_frag_h(qg, HD, qr, 32, hf);

        // ---- scores S = (q*scale) @ K^T + bias, masked ----
        for (int kt = 0; kt < 18; ++kt) {
            v16h kb0 = load_frag_h(Ksm, 72, kt * 16 + r, 0, hf);
            v16h kb1 = load_frag_h(Ksm, 72, kt * 16 + r, 32, hf);
            v8f s = zero8f();
            s = wmma16(qa0, kb0, s);
            s = wmma16(qa1, kb1, s);
            const int kj = kt * 16 + r;
#pragma unroll
            for (int e = 0; e < 8; ++e) {
                const int qi = qt * 16 + e + 8 * hf;
                float v = s[e];
                if (kj < NSEQ) {
                    if (qi < NSEQ) v += biasH[qi * NSEQ + kj];
                    if (!mb[kj]) v = -1e30f;
                } else {
                    v = -1e30f;
                }
                sc[(e + 8 * hf) * NPAD + kj] = v;
            }
        }

        // ---- row softmax (wave-cooperative over 288 cols) ----
        for (int rr = 0; rr < 16; ++rr) {
            float* rowp = sc + rr * NPAD;
            float mx = -1e30f;
            for (int c = lane; c < NPAD; c += 32) mx = fmaxf(mx, rowp[c]);
#pragma unroll
            for (int o = 16; o; o >>= 1) mx = fmaxf(mx, __shfl_xor(mx, o));
            float sum = 0.f;
            for (int c = lane; c < NPAD; c += 32) {
                const float ev = __expf(rowp[c] - mx);
                rowp[c] = ev;
                sum += ev;
            }
#pragma unroll
            for (int o = 16; o; o >>= 1) sum += __shfl_xor(sum, o);
            const float inv = sum > 0.f ? 1.f / sum : 0.f;
            for (int c = lane; c < NPAD; c += 32) rowp[c] *= inv;
        }

        // ---- out = P @ V ----
        v8f o0 = zero8f(), o1 = zero8f(), o2 = zero8f(), o3 = zero8f();
        for (int kk = 0; kk < 9; ++kk) {
            const float* pp = sc + r * NPAD + kk * 32 + hf * 8;
            v16h pa;
#pragma unroll
            for (int i = 0; i < 8; ++i) {
                pa[i] = (_Float16)pp[i];
                pa[i + 8] = (_Float16)pp[i + 16];
            }
            v16h vb0 = load_frag_h(Vtm, 296, r, kk * 32, hf);
            v16h vb1 = load_frag_h(Vtm, 296, 16 + r, kk * 32, hf);
            v16h vb2 = load_frag_h(Vtm, 296, 32 + r, kk * 32, hf);
            v16h vb3 = load_frag_h(Vtm, 296, 48 + r, kk * 32, hf);
            o0 = wmma16(pa, vb0, o0);
            o1 = wmma16(pa, vb1, o1);
            o2 = wmma16(pa, vb2, o2);
            o3 = wmma16(pa, vb3, o3);
        }
#pragma unroll
        for (int e = 0; e < 8; ++e) {
            const int qi = qt * 16 + e + 8 * hf;
            if (qi < NSEQ) {
                _Float16* op = ao + ((long)(b * NSEQ + qi)) * DIMC + h * HD + r;
                op[0] = (_Float16)o0[e];
                op[16] = (_Float16)o1[e];
                op[32] = (_Float16)o2[e];
                op[48] = (_Float16)o3[e];
            }
        }
    }
}

// ---------------------------------------------------------------------------
// Kernel 3: out = attn_out(f16,[B*N,1024]) @ proj_w^T + proj_b -> f32 d_out
// A tile is already f16 -> stage via async global->LDS DMA (no VGPR roundtrip)
// ---------------------------------------------------------------------------
__global__ __launch_bounds__(256) void proj_gemm_kernel(
    const _Float16* __restrict__ ain, const float* __restrict__ w,
    const float* __restrict__ pb, float* __restrict__ out) {
    __shared__ _Float16 As[128 * 40];
    __shared__ _Float16 Bs[64 * 40];

    const int tid = threadIdx.x;
    const int lane = tid & 31, wv = tid >> 5;
    const int wm = wv & 3, wn = wv >> 2;
    const int r = lane & 15, hf = lane >> 4;
    const int j0 = blockIdx.x * 64;
    const int m0 = blockIdx.y * 128;

    v8f acc00 = zero8f(), acc01 = zero8f(), acc10 = zero8f(), acc11 = zero8f();

    const int xr = tid >> 1, xc = (tid & 1) * 16;
    const int wr = tid >> 2, wc = (tid & 3) * 8;
    const int m = m0 + xr;
    const unsigned int ldsA = (unsigned int)(uintptr_t)&As[xr * 40 + xc];

    for (int kk0 = 0; kk0 < DIMC; kk0 += 32) {
        if (m < M_TOTAL) {
            // async copy 32B (two b128, IOFFSET applies to both LDS and mem)
            const unsigned long long ga =
                (unsigned long long)(uintptr_t)(ain + (long)m * DIMC + kk0 + xc);
            asm volatile("global_load_async_to_lds_b128 %0, %1, off"
                         :: "v"(ldsA), "v"(ga) : "memory");
            asm volatile("global_load_async_to_lds_b128 %0, %1, off offset:16"
                         :: "v"(ldsA), "v"(ga) : "memory");
        } else {
            v8h z;
#pragma unroll
            for (int i = 0; i < 8; ++i) z[i] = (_Float16)0.f;
            *(v8h*)&As[xr * 40 + xc] = z;
            *(v8h*)&As[xr * 40 + xc + 8] = z;
        }
        {
            const float* p = w + (long)(j0 + wr) * DIMC + kk0 + wc;
            v8h hw;
#pragma unroll
            for (int i = 0; i < 8; ++i) hw[i] = (_Float16)p[i];
            *(v8h*)&Bs[wr * 40 + wc] = hw;
        }
        asm volatile("s_wait_asynccnt 0x0" ::: "memory");
        __syncthreads();

        v16h a0 = load_frag_h(As, 40, wm * 32 + r, 0, hf);
        v16h a1 = load_frag_h(As, 40, wm * 32 + 16 + r, 0, hf);
        v16h b0 = load_frag_h(Bs, 40, wn * 32 + r, 0, hf);
        v16h b1 = load_frag_h(Bs, 40, wn * 32 + 16 + r, 0, hf);
        acc00 = wmma16(a0, b0, acc00);
        acc01 = wmma16(a0, b1, acc01);
        acc10 = wmma16(a1, b0, acc10);
        acc11 = wmma16(a1, b1, acc11);
        __syncthreads();
    }

#pragma unroll
    for (int mi = 0; mi < 2; ++mi) {
#pragma unroll
        for (int ni = 0; ni < 2; ++ni) {
            v8f av = mi == 0 ? (ni == 0 ? acc00 : acc01)
                             : (ni == 0 ? acc10 : acc11);
            const int jj = j0 + wn * 32 + ni * 16 + r;
            const float bias = pb[jj];
#pragma unroll
            for (int e = 0; e < 8; ++e) {
                const int mm = m0 + wm * 32 + mi * 16 + e + 8 * hf;
                if (mm < M_TOTAL) out[(long)mm * DIMC + jj] = av[e] + bias;
            }
        }
    }
}

// ---------------------------------------------------------------------------
extern "C" void kernel_launch(void* const* d_in, const int* in_sizes, int n_in,
                              void* d_out, int out_size, void* d_ws,
                              size_t ws_size, hipStream_t stream) {
    (void)in_sizes; (void)n_in; (void)out_size; (void)ws_size;
    const float* x = (const float*)d_in[0];
    const float* qkv_w = (const float*)d_in[1];
    const float* q_bias = (const float*)d_in[2];
    const float* v_bias = (const float*)d_in[3];
    const float* rel_tab = (const float*)d_in[4];
    const float* proj_w = (const float*)d_in[5];
    const float* proj_b = (const float*)d_in[6];
    const int* rel_idx = (const int*)d_in[7];
    const unsigned char* mask = (const unsigned char*)d_in[8];
    float* out = (float*)d_out;

    char* ws = (char*)d_ws;
    const size_t slab = (size_t)BATCH * HEADS * NSEQ * HD * sizeof(_Float16);  // 33,685,504 B
    _Float16* qs = (_Float16*)(ws);
    _Float16* ks = (_Float16*)(ws + slab);
    _Float16* vs = (_Float16*)(ws + 2 * slab);
    _Float16* ao = (_Float16*)(ws + 3 * slab);
    float* biasTab = (float*)(ws + 4 * slab);  // + 4,227,136 B

    // 0) expand relative-position bias to [H,N,N]
    bias_expand_kernel<<<(NSEQ * NSEQ + 255) / 256, 256, 0, stream>>>(
        rel_tab, rel_idx, biasTab);

    // 1) qkv projection (q pre-scaled)
    dim3 g1(3 * DIMC / 64, (M_TOTAL + 127) / 128);  // 48 x 129
    qkv_gemm_kernel<<<g1, 256, 0, stream>>>(x, qkv_w, q_bias, v_bias, qs, ks, vs);

    // 2) fused attention, one block per (b,h); 226,816 B dynamic LDS
    attn_kernel<<<BATCH * HEADS, 256, 226816, stream>>>(qs, ks, vs, biasTab, mask, ao);

    // 3) output projection
    dim3 g3(DIMC / 64, (M_TOTAL + 127) / 128);  // 16 x 129
    proj_gemm_kernel<<<g3, 256, 0, stream>>>(ao, proj_w, proj_b, out);
}
